// SpectralMamba_10788957847757
// MI455X (gfx1250) — matrix operand order repaired and checked by reading, not compiled, not run on hardware
//
#include <hip/hip_runtime.h>
#include <hip/hip_bf16.h>
#include <math.h>

// ---- problem constants (match reference) ----
#define C_IN   96
#define D2     192          // d
#define D4     768          // 4d
#define NSTATE 16
#define DR     6
#define NJ     (DR + 2*NSTATE)   // 38
#define NJP    48                // NJ padded to 3 WMMA n-tiles
#define BB     2
#define HH     96
#define WW     96
#define LL     (HH*WW)           // 9216
#define NROWS  (BB*LL)           // 18432
#define EOUT   (2*D2)            // 384
#define NCH    96                // scan chunks
#define CLEN   96                // steps per chunk (NCH*CLEN == LL)

typedef __attribute__((ext_vector_type(16))) _Float16 v16h;
typedef __attribute__((ext_vector_type(8)))  _Float16 v8h;
typedef __attribute__((ext_vector_type(8)))  float    v8f;

// ------------------------------------------------------------------
// Pack kernels
__global__ void pack_a_f16(const float* __restrict__ x, _Float16* __restrict__ A16) {
  int id = blockIdx.x * blockDim.x + threadIdx.x;
  if (id >= NROWS * C_IN) return;
  int row = id / C_IN, c = id % C_IN;
  int b = row / LL, t = row % LL;
  A16[id] = (_Float16)x[((size_t)(b * C_IN + c)) * LL + t];
}

__global__ void pack_b_f16(const float* __restrict__ Win, _Float16* __restrict__ Bp) {
  int id = blockIdx.x * blockDim.x + threadIdx.x;
  if (id >= EOUT * C_IN) return;
  int n = id / C_IN, k = id % C_IN;
  Bp[id] = (_Float16)Win[(size_t)k * EOUT + n];
}

// fold W_x (768,38) into fwd/rev halves, packed f16 [j][c] with j padded to 48
__global__ void pack_wx16(const float* __restrict__ Wx, _Float16* __restrict__ Wxfp,
                          _Float16* __restrict__ Wxrp) {
  int id = blockIdx.x * blockDim.x + threadIdx.x;
  if (id >= NJP * D2) return;
  int j = id / D2, c = id % D2;
  float f = 0.f, r = 0.f;
  if (j < NJ) {
    f = Wx[(size_t)c * NJ + j]            + Wx[(size_t)(c + D2) * NJ + j];
    r = Wx[(size_t)(c + 2 * D2) * NJ + j] + Wx[(size_t)(c + 3 * D2) * NJ + j];
  }
  Wxfp[id] = (_Float16)f;
  Wxrp[id] = (_Float16)r;
}

// W_out (192,96) -> f16 [n][k] = (96,192)
__global__ void pack_wout16(const float* __restrict__ Wout, _Float16* __restrict__ Wp) {
  int id = blockIdx.x * blockDim.x + threadIdx.x;
  if (id >= C_IN * D2) return;
  int n = id / D2, k = id % D2;
  Wp[id] = (_Float16)Wout[(size_t)k * C_IN + n];
}

// ------------------------------------------------------------------
// K1: xz = A16 @ W_in + b_in  (M=18432, N=384, K=96) via WMMA
__global__ void __launch_bounds__(256)
gemm_in_wmma(const _Float16* __restrict__ A, const _Float16* __restrict__ Bp,
             const float* __restrict__ bias, float* __restrict__ C) {
  const int lane = threadIdx.x & 31;
  const int wave = threadIdx.x >> 5;
  const int tile = blockIdx.x * 8 + wave;
  const int mt = tile / (EOUT / 16);
  const int nt = tile % (EOUT / 16);
  const int r  = lane & 15;
  const int hi = lane >> 4;
  const _Float16* arow = A  + (size_t)(mt * 16 + r) * C_IN + hi * 8;
  const _Float16* brow = Bp + (size_t)(nt * 16 + r) * C_IN + hi * 8;
  v8f acc = {};
#pragma unroll
  for (int k0 = 0; k0 < C_IN; k0 += 32) {
    if (k0 + 32 < C_IN) {
      __builtin_prefetch(arow + k0 + 32, 0, 3);
      __builtin_prefetch(brow + k0 + 32, 0, 3);
    }
    v8h alo = *(const v8h*)(arow + k0);
    v8h ahi = *(const v8h*)(arow + k0 + 16);
    v8h blo = *(const v8h*)(brow + k0);
    v8h bhi = *(const v8h*)(brow + k0 + 16);
    v16h a = __builtin_shufflevector(alo, ahi, 0,1,2,3,4,5,6,7,8,9,10,11,12,13,14,15);
    v16h b = __builtin_shufflevector(blo, bhi, 0,1,2,3,4,5,6,7,8,9,10,11,12,13,14,15);
    acc = __builtin_amdgcn_wmma_f32_16x16x32_f16(false, a, false, b, (short)0, acc, false, false);
  }
  const int col = nt * 16 + r;
  const int m0  = mt * 16 + hi * 8;
  const float bv = bias[col];
#pragma unroll
  for (int v = 0; v < 8; ++v)
    C[(size_t)(m0 + v) * EOUT + col] = acc[v] + bv;
}

// ------------------------------------------------------------------
// K2: depthwise 3x3 conv + bias + SiLU; emits f32 (for scan) and f16 (for WMMA A-frags)
__global__ void dwconv_silu(const float* __restrict__ xz, const float* __restrict__ wdw,
                            const float* __restrict__ bdw, float* __restrict__ xc,
                            _Float16* __restrict__ xc16) {
  int id = blockIdx.x * blockDim.x + threadIdx.x;
  if (id >= NROWS * D2) return;
  int c = id % D2;
  int row = id / D2;            // b*L + t
  int t = row % LL, b = row / LL;
  int h = t / WW, w = t % WW;
  float acc = bdw[c];
#pragma unroll
  for (int dy = 0; dy < 3; ++dy) {
    int hh = h + dy - 1;
    if ((unsigned)hh >= HH) continue;
#pragma unroll
    for (int dx = 0; dx < 3; ++dx) {
      int ww2 = w + dx - 1;
      if ((unsigned)ww2 >= WW) continue;
      acc += xz[((size_t)b * LL + hh * WW + ww2) * EOUT + c] * wdw[c * 9 + dy * 3 + dx];
    }
  }
  float s = acc / (1.0f + __expf(-acc));
  xc[(size_t)row * D2 + c] = s;
  xc16[(size_t)row * D2 + c] = (_Float16)s;
}

// ------------------------------------------------------------------
// K3: x_dbl = Xf @ Wxf^T + Xr @ Wxr^T via WMMA (M=18432, N=48, K=192), split to dtr/Bm/Cm
__global__ void __launch_bounds__(256)
xdbl_wmma(const _Float16* __restrict__ X, const _Float16* __restrict__ Wf,
          const _Float16* __restrict__ Wr, float* __restrict__ dtr,
          float* __restrict__ Bm, float* __restrict__ Cm) {
  const int lane = threadIdx.x & 31;
  const int wave = threadIdx.x >> 5;
  const int tile = blockIdx.x * 8 + wave;
  const int mt = tile / (NJP / 16);
  const int nt = tile % (NJP / 16);
  const int r  = lane & 15;
  const int hi = lane >> 4;
  const int rowg   = mt * 16 + r;
  const int bb     = rowg / LL;
  const int tt     = rowg - bb * LL;
  const int rowrev = bb * LL + (LL - 1 - tt);
  const _Float16* arf = X  + (size_t)rowg   * D2 + hi * 8;
  const _Float16* arr = X  + (size_t)rowrev * D2 + hi * 8;
  const _Float16* bf  = Wf + (size_t)(nt * 16 + r) * D2 + hi * 8;
  const _Float16* br  = Wr + (size_t)(nt * 16 + r) * D2 + hi * 8;
  v8f acc = {};
#pragma unroll
  for (int k0 = 0; k0 < D2; k0 += 32) {
    if (k0 + 32 < D2) {
      __builtin_prefetch(arf + k0 + 32, 0, 3);
      __builtin_prefetch(arr + k0 + 32, 0, 3);
    }
    v8h aflo = *(const v8h*)(arf + k0);
    v8h afhi = *(const v8h*)(arf + k0 + 16);
    v8h bflo = *(const v8h*)(bf + k0);
    v8h bfhi = *(const v8h*)(bf + k0 + 16);
    v16h af = __builtin_shufflevector(aflo, afhi, 0,1,2,3,4,5,6,7,8,9,10,11,12,13,14,15);
    v16h wf = __builtin_shufflevector(bflo, bfhi, 0,1,2,3,4,5,6,7,8,9,10,11,12,13,14,15);
    acc = __builtin_amdgcn_wmma_f32_16x16x32_f16(false, af, false, wf, (short)0, acc, false, false);
    v8h arlo = *(const v8h*)(arr + k0);
    v8h arhi = *(const v8h*)(arr + k0 + 16);
    v8h brlo = *(const v8h*)(br + k0);
    v8h brhi = *(const v8h*)(br + k0 + 16);
    v16h ar = __builtin_shufflevector(arlo, arhi, 0,1,2,3,4,5,6,7,8,9,10,11,12,13,14,15);
    v16h wr = __builtin_shufflevector(brlo, brhi, 0,1,2,3,4,5,6,7,8,9,10,11,12,13,14,15);
    acc = __builtin_amdgcn_wmma_f32_16x16x32_f16(false, ar, false, wr, (short)0, acc, false, false);
  }
  const int j  = nt * 16 + r;          // 0..47; only 0..37 are live
  const int m0 = mt * 16 + hi * 8;
#pragma unroll
  for (int v = 0; v < 8; ++v) {
    const size_t row = (size_t)(m0 + v);
    if (j < DR)               dtr[row * DR + j] = acc[v];
    else if (j < DR + NSTATE) Bm[row * NSTATE + (j - DR)] = acc[v];
    else if (j < NJ)          Cm[row * NSTATE + (j - DR - NSTATE)] = acc[v];
  }
}

// K4: delta = softplus(dtr @ W_delta + b_delta)
__global__ void delta_kernel(const float* __restrict__ dtr, const float* __restrict__ Wd,
                             const float* __restrict__ bd, float* __restrict__ delta) {
  int id = blockIdx.x * blockDim.x + threadIdx.x;
  if (id >= NROWS * D4) return;
  int dch = id % D4, row = id / D4;
  float s = bd[dch];
#pragma unroll
  for (int rr = 0; rr < DR; ++rr)
    s += dtr[(size_t)row * DR + rr] * Wd[rr * D4 + dch];
  delta[(size_t)row * D4 + dch] = fmaxf(s, 0.f) + log1pf(__expf(-fabsf(s)));
}

// ------------------------------------------------------------------
// Chunked selective scan: h_t = dA_t*h + du_t*B_t is elementwise-affine, so chunks compose.
// Phase A: local scan per chunk from h=0; record P=prod(dA), q=h_end. 4608 waves.
__global__ void scan_local(const float* __restrict__ xc, const float* __restrict__ delta,
                           const float* __restrict__ Bm, const float* __restrict__ Cm,
                           const float* __restrict__ A_log, const float* __restrict__ Dp,
                           float* __restrict__ yacc, float* __restrict__ Pbuf,
                           float* __restrict__ Qbuf) {
  int blk = blockIdx.x;
  const int chunk = blk % NCH; blk /= NCH;
  const int cg = blk % 6;      blk /= 6;
  const int dir = blk % 4;
  const int b = blk / 4;
  const int c = cg * 32 + threadIdx.x;
  const int dch = dir * D2 + c;
  float Ar[NSTATE], h[NSTATE], P[NSTATE];
#pragma unroll
  for (int n = 0; n < NSTATE; ++n) {
    Ar[n] = -__expf(A_log[dch * NSTATE + n]);
    h[n] = 0.f; P[n] = 1.f;
  }
  const float dpar = Dp[dch];
  const size_t bL = (size_t)b * LL;
  const int t0 = chunk * CLEN;
  for (int t = t0; t < t0 + CLEN; ++t) {
    const int tu   = (dir < 2) ? t : (LL - 1 - t);
    const float u  = xc[(bL + tu) * D2 + c];
    const float dt = delta[(bL + t) * D4 + dch];
    const float du = dt * u;
    const float* bp = Bm + (bL + t) * NSTATE;
    const float* cp = Cm + (bL + t) * NSTATE;
    float y = dpar * u;
#pragma unroll
    for (int n = 0; n < NSTATE; ++n) {
      float dA = __expf(dt * Ar[n]);
      P[n] *= dA;
      float hn = dA * h[n] + du * bp[n];
      h[n] = hn;
      y += hn * cp[n];
    }
    atomicAdd(&yacc[(bL + tu) * D2 + c], y);
  }
  const size_t sbase = (((size_t)b * D4 + dch) * NCH + chunk) * NSTATE;
#pragma unroll
  for (int n = 0; n < NSTATE; ++n) { Pbuf[sbase + n] = P[n]; Qbuf[sbase + n] = h[n]; }
}

// Phase B: sequential prefix over chunk summaries; store entering state per chunk.
__global__ void scan_prefix(const float* __restrict__ Pbuf, const float* __restrict__ Qbuf,
                            float* __restrict__ h0s) {
  const int gi = blockIdx.x * 32 + threadIdx.x;   // 0..1535 = b*768 + dch
  float h[NSTATE];
#pragma unroll
  for (int n = 0; n < NSTATE; ++n) h[n] = 0.f;
  const size_t base = (size_t)gi * NCH * NSTATE;
  for (int k = 0; k < NCH; ++k) {
    const size_t s = base + (size_t)k * NSTATE;
#pragma unroll
    for (int n = 0; n < NSTATE; ++n) {
      h0s[s + n] = h[n];
      h[n] = Pbuf[s + n] * h[n] + Qbuf[s + n];
    }
  }
}

// Phase C: y_t += C_t . (prefixprod(dA) o H0) for each chunk.
__global__ void scan_fix(const float* __restrict__ delta, const float* __restrict__ Cm,
                         const float* __restrict__ A_log, const float* __restrict__ h0s,
                         float* __restrict__ yacc) {
  int blk = blockIdx.x;
  const int chunk = blk % NCH; blk /= NCH;
  if (chunk == 0) return;                    // entering state is zero
  const int cg = blk % 6;      blk /= 6;
  const int dir = blk % 4;
  const int b = blk / 4;
  const int c = cg * 32 + threadIdx.x;
  const int dch = dir * D2 + c;
  float Ar[NSTATE], hh[NSTATE];
  const size_t sbase = (((size_t)b * D4 + dch) * NCH + chunk) * NSTATE;
#pragma unroll
  for (int n = 0; n < NSTATE; ++n) {
    Ar[n] = -__expf(A_log[dch * NSTATE + n]);
    hh[n] = h0s[sbase + n];
  }
  const size_t bL = (size_t)b * LL;
  const int t0 = chunk * CLEN;
  for (int t = t0; t < t0 + CLEN; ++t) {
    const int tu   = (dir < 2) ? t : (LL - 1 - t);
    const float dt = delta[(bL + t) * D4 + dch];
    const float* cp = Cm + (bL + t) * NSTATE;
    float y = 0.f;
#pragma unroll
    for (int n = 0; n < NSTATE; ++n) {
      hh[n] *= __expf(dt * Ar[n]);
      y += hh[n] * cp[n];
    }
    atomicAdd(&yacc[(bL + tu) * D2 + c], y);
  }
}

// ------------------------------------------------------------------
// K6a: LayerNorm(192) + * SiLU(z) -> f16 s-matrix (row, 192). One wave per row.
__global__ void __launch_bounds__(256)
ln_silu(const float* __restrict__ yacc, const float* __restrict__ xz,
        const float* __restrict__ ln_g, const float* __restrict__ ln_b,
        _Float16* __restrict__ s16) {
  const int lane = threadIdx.x & 31;
  const int wave = threadIdx.x >> 5;
  const int row  = blockIdx.x * 8 + wave;   // grid exact: NROWS/8
  float v[6], sum = 0.f;
#pragma unroll
  for (int j = 0; j < 6; ++j) { v[j] = yacc[(size_t)row * D2 + lane + 32 * j]; sum += v[j]; }
#pragma unroll
  for (int off = 16; off > 0; off >>= 1) sum += __shfl_xor(sum, off, 32);
  const float mu = sum * (1.0f / D2);
  float vs = 0.f;
#pragma unroll
  for (int j = 0; j < 6; ++j) { float d = v[j] - mu; vs += d * d; }
#pragma unroll
  for (int off = 16; off > 0; off >>= 1) vs += __shfl_xor(vs, off, 32);
  const float rs = rsqrtf(vs * (1.0f / D2) + 1e-5f);
#pragma unroll
  for (int j = 0; j < 6; ++j) {
    int cc = lane + 32 * j;
    float yn = (v[j] - mu) * rs * ln_g[cc] + ln_b[cc];
    float z  = xz[(size_t)row * EOUT + D2 + cc];
    s16[(size_t)row * D2 + cc] = (_Float16)(yn * (z / (1.0f + __expf(-z))));
  }
}

// K6b: out = s16 @ W_out (M=18432, N=96, K=192) via WMMA; store transposed (b, cout, t)
__global__ void __launch_bounds__(256)
gemm_out_wmma(const _Float16* __restrict__ A, const _Float16* __restrict__ Bp,
              float* __restrict__ out) {
  const int lane = threadIdx.x & 31;
  const int wave = threadIdx.x >> 5;
  const int tile = blockIdx.x * 8 + wave;
  const int mt = tile / (C_IN / 16);
  const int nt = tile % (C_IN / 16);
  const int r  = lane & 15;
  const int hi = lane >> 4;
  const _Float16* arow = A  + (size_t)(mt * 16 + r) * D2 + hi * 8;
  const _Float16* brow = Bp + (size_t)(nt * 16 + r) * D2 + hi * 8;
  v8f acc = {};
#pragma unroll
  for (int k0 = 0; k0 < D2; k0 += 32) {
    if (k0 + 32 < D2) {
      __builtin_prefetch(arow + k0 + 32, 0, 3);
      __builtin_prefetch(brow + k0 + 32, 0, 3);
    }
    v8h alo = *(const v8h*)(arow + k0);
    v8h ahi = *(const v8h*)(arow + k0 + 16);
    v8h blo = *(const v8h*)(brow + k0);
    v8h bhi = *(const v8h*)(brow + k0 + 16);
    v16h a = __builtin_shufflevector(alo, ahi, 0,1,2,3,4,5,6,7,8,9,10,11,12,13,14,15);
    v16h b = __builtin_shufflevector(blo, bhi, 0,1,2,3,4,5,6,7,8,9,10,11,12,13,14,15);
    acc = __builtin_amdgcn_wmma_f32_16x16x32_f16(false, a, false, b, (short)0, acc, false, false);
  }
  const int cout  = nt * 16 + r;
  const int m0    = mt * 16 + hi * 8;
  const int b     = (mt * 16) / LL;         // tiles never straddle b (LL % 16 == 0)
  const int tbase = m0 - b * LL;
  float* op = out + ((size_t)b * C_IN + cout) * LL + tbase;
#pragma unroll
  for (int v = 0; v < 8; ++v) op[v] = acc[v];   // 8 consecutive t -> coalescable
}

// ------------------------------------------------------------------
extern "C" void kernel_launch(void* const* d_in, const int* in_sizes, int n_in,
                              void* d_out, int out_size, void* d_ws, size_t ws_size,
                              hipStream_t stream) {
  (void)in_sizes; (void)n_in; (void)out_size; (void)ws_size;
  const float* x       = (const float*)d_in[0];
  const float* W_in    = (const float*)d_in[1];
  const float* b_in    = (const float*)d_in[2];
  const float* w_dw    = (const float*)d_in[3];
  const float* b_dw    = (const float*)d_in[4];
  const float* A_log   = (const float*)d_in[5];
  const float* Dp      = (const float*)d_in[6];
  const float* W_x     = (const float*)d_in[7];
  const float* W_delta = (const float*)d_in[8];
  const float* b_delta = (const float*)d_in[9];
  const float* ln_g    = (const float*)d_in[10];
  const float* ln_b    = (const float*)d_in[11];
  const float* W_out   = (const float*)d_in[12];
  float* out = (float*)d_out;

  char* ws = (char*)d_ws;
  size_t off = 0;
  auto carve = [&](size_t bytes) -> char* {
    char* p = ws + off;
    off += (bytes + 255) & ~(size_t)255;
    return p;
  };
  _Float16* A16   = (_Float16*)carve((size_t)NROWS * C_IN * sizeof(_Float16));
  _Float16* Bp    = (_Float16*)carve((size_t)EOUT * C_IN * sizeof(_Float16));
  _Float16* Wxfp  = (_Float16*)carve((size_t)NJP * D2 * sizeof(_Float16));
  _Float16* Wxrp  = (_Float16*)carve((size_t)NJP * D2 * sizeof(_Float16));
  _Float16* Wo16  = (_Float16*)carve((size_t)C_IN * D2 * sizeof(_Float16));
  _Float16* xc16  = (_Float16*)carve((size_t)NROWS * D2 * sizeof(_Float16));
  _Float16* s16   = (_Float16*)carve((size_t)NROWS * D2 * sizeof(_Float16));
  float* xz       = (float*)carve((size_t)NROWS * EOUT * sizeof(float));
  float* xc       = (float*)carve((size_t)NROWS * D2 * sizeof(float));
  float* dtr      = (float*)carve((size_t)NROWS * DR * sizeof(float));
  float* Bmw      = (float*)carve((size_t)NROWS * NSTATE * sizeof(float));
  float* Cmw      = (float*)carve((size_t)NROWS * NSTATE * sizeof(float));
  float* delta    = (float*)carve((size_t)NROWS * D4 * sizeof(float));
  float* yacc     = (float*)carve((size_t)NROWS * D2 * sizeof(float));
  float* Pbuf     = (float*)carve((size_t)BB * D4 * NCH * NSTATE * sizeof(float));
  float* Qbuf     = (float*)carve((size_t)BB * D4 * NCH * NSTATE * sizeof(float));
  float* h0s      = (float*)carve((size_t)BB * D4 * NCH * NSTATE * sizeof(float));

  pack_a_f16<<<(NROWS * C_IN + 255) / 256, 256, 0, stream>>>(x, A16);
  pack_b_f16<<<(EOUT * C_IN + 255) / 256, 256, 0, stream>>>(W_in, Bp);
  pack_wx16<<<(NJP * D2 + 255) / 256, 256, 0, stream>>>(W_x, Wxfp, Wxrp);
  pack_wout16<<<(C_IN * D2 + 255) / 256, 256, 0, stream>>>(W_out, Wo16);

  gemm_in_wmma<<<(NROWS / 16) * (EOUT / 16) / 8, 256, 0, stream>>>(A16, Bp, b_in, xz);
  dwconv_silu<<<(NROWS * D2 + 255) / 256, 256, 0, stream>>>(xz, w_dw, b_dw, xc, xc16);

  xdbl_wmma<<<(NROWS / 16) * (NJP / 16) / 8, 256, 0, stream>>>(xc16, Wxfp, Wxrp, dtr, Bmw, Cmw);
  delta_kernel<<<(NROWS * D4 + 255) / 256, 256, 0, stream>>>(dtr, W_delta, b_delta, delta);

  hipMemsetAsync(yacc, 0, (size_t)NROWS * D2 * sizeof(float), stream);
  const int scan_blocks = BB * 4 * (D2 / 32) * NCH;   // 4608
  scan_local<<<scan_blocks, 32, 0, stream>>>(xc, delta, Bmw, Cmw, A_log, Dp, yacc, Pbuf, Qbuf);
  scan_prefix<<<(BB * D4) / 32, 32, 0, stream>>>(Pbuf, Qbuf, h0s);
  scan_fix<<<scan_blocks, 32, 0, stream>>>(delta, Cmw, A_log, h0s, yacc);

  ln_silu<<<NROWS / 8, 256, 0, stream>>>(yacc, xz, ln_g, ln_b, s16);
  gemm_out_wmma<<<(NROWS / 16) * (C_IN / 16) / 8, 256, 0, stream>>>(s16, Wo16, out);
}
